// SpatioTemporalBlock_89781996356033
// MI455X (gfx1250) — matrix operand order repaired and checked by reading, not compiled, not run on hardware
//
#include <hip/hip_runtime.h>
#include <hip/hip_bf16.h>
#include <math.h>

#define B_    8
#define C_    64
#define NP_   512
#define D_    256
#define D2_   512
#define TOPK_ 8
#define TAU_  0.2f
#define ROWS_ 64            // rows per MLP block tile
#define NT_   512           // threads per MLP block (16 waves)
#define XPAD_ 264           // padded xs row (256+8): 16B aligned, conflict-free
#define HPAD_ 520           // padded hs row (512+8)
#define WPAD_ 40            // padded wp row (32+8): byte stride 80, 16B aligned

typedef __attribute__((ext_vector_type(16))) __bf16 v16bf;
typedef __attribute__((ext_vector_type(8)))  float  v8f;

union Frag { v16bf v; uint4 q[2]; };

// ---------------- K0: convert + transpose W1/W2 to bf16 N-major (NxK) ----------------
__global__ void k_cvt(const float* __restrict__ W1, const float* __restrict__ W2,
                      __bf16* __restrict__ W1T, __bf16* __restrict__ W2T) {
    int i = blockIdx.x * blockDim.x + threadIdx.x;
    if (i < D_ * D2_) {                 // W1T[n][k], n<512, k<256
        int n = i / D_, k = i % D_;
        W1T[i] = (__bf16)W1[k * D2_ + n];
    }
    if (i < D2_ * D_) {                 // W2T[n][k], n<256, k<512
        int n = i / D2_, k = i % D2_;
        W2T[i] = (__bf16)W2[k * D_ + n];
    }
}

// ---------------- K1: Hc = mean over Np ----------------
__global__ void k_mean(const float* __restrict__ H, float* __restrict__ Hc) {
    int bc = blockIdx.x;
    int d  = threadIdx.x;
    const float* p = H + (size_t)bc * NP_ * D_ + d;
    float s = 0.f;
    for (int n = 0; n < NP_; ++n) s += p[(size_t)n * D_];
    Hc[bc * D_ + d] = s * (1.0f / NP_);
}

// ---------------- K2a: Q = Hc@Wq+bq ; K = Hc@Wk+bk ----------------
__global__ void k_qk(const float* __restrict__ Hc,
                     const float* __restrict__ Wq, const float* __restrict__ bq,
                     const float* __restrict__ Wk, const float* __restrict__ bk,
                     float* __restrict__ Q, float* __restrict__ K) {
    int row = blockIdx.x;
    int d   = threadIdx.x;
    __shared__ float hrow[D_];
    hrow[d] = Hc[row * D_ + d];
    __syncthreads();
    float q = bq[d], k = bk[d];
    for (int e = 0; e < D_; ++e) {
        float h = hrow[e];
        q += h * Wq[e * D_ + d];
        k += h * Wk[e * D_ + d];
    }
    Q[row * D_ + d] = q;
    K[row * D_ + d] = k;
}

// ---------------- K2b: per-batch softmax + top-8 sparsify ----------------
__global__ void k_attn(const float* __restrict__ Q, const float* __restrict__ K,
                       float* __restrict__ w8, int* __restrict__ i8) {
    int b = blockIdx.x;
    int t = threadIdx.x;
    __shared__ float S[C_][C_];
    for (int e = t; e < C_ * C_; e += 256) {
        int i = e / C_, j = e % C_;
        const float* qp = Q + (size_t)(b * C_ + i) * D_;
        const float* kp = K + (size_t)(b * C_ + j) * D_;
        float s = 0.f;
        for (int d = 0; d < D_; ++d) s += qp[d] * kp[d];
        S[i][j] = s * (1.0f / 16.0f) * (1.0f / TAU_);
    }
    __syncthreads();
    if (t < C_) {
        int i = t;
        float m = -INFINITY;
        for (int j = 0; j < C_; ++j) m = fmaxf(m, S[i][j]);
        float sum = 0.f;
        for (int j = 0; j < C_; ++j) { float e = expf(S[i][j] - m); S[i][j] = e; sum += e; }
        float inv = 1.0f / sum;
        int   sel[TOPK_];
        float val[TOPK_];
        float ssum = 0.f;
        for (int p = 0; p < TOPK_; ++p) {
            float best = -1.f; int bj = 0;
            for (int j = 0; j < C_; ++j) {
                bool used = false;
                for (int q2 = 0; q2 < p; ++q2) if (sel[q2] == j) used = true;
                float a = S[i][j];
                if (!used && a > best) { best = a; bj = j; }
            }
            sel[p] = bj;
            val[p] = best * inv;
            ssum  += best * inv;
        }
        float r = 1.0f / (ssum + 1e-6f);
        for (int p = 0; p < TOPK_; ++p) {
            w8[(b * C_ + i) * TOPK_ + p] = val[p] * r;
            i8[(b * C_ + i) * TOPK_ + p] = sel[p];
        }
    }
}

// ---------------- K3: sparse mix + residual + LayerNorm (LDS slab) ----------------
__global__ void k_mix_ln(const float* __restrict__ H,
                         const float* __restrict__ w8, const int* __restrict__ i8,
                         const float* __restrict__ g, const float* __restrict__ bb,
                         float* __restrict__ H1out, __bf16* __restrict__ X) {
    int bn = blockIdx.x;
    int b  = bn / NP_;
    int n  = bn % NP_;
    int d  = threadIdx.x;
    int lane = d & 31, wave = d >> 5;
    __shared__ float slab[C_][D_];    // 64 KB: H[b, :, n, :]
    __shared__ float red[2][8];
    __shared__ float wl[C_][TOPK_];
    __shared__ int   il[C_][TOPK_];
    for (int e = d; e < C_ * TOPK_; e += 256) {
        wl[e / TOPK_][e % TOPK_] = w8[b * C_ * TOPK_ + e];
        il[e / TOPK_][e % TOPK_] = i8[b * C_ * TOPK_ + e];
    }
    { // stage slab: 64 rows x 1KB, coalesced uint4
        const float* src = H + (size_t)b * C_ * NP_ * D_ + (size_t)n * D_;
        for (int idx = d; idx < C_ * (D_ / 4); idx += 256) {
            int c = idx >> 6, j = idx & 63;
            ((uint4*)&slab[c][0])[j] = ((const uint4*)(src + (size_t)c * NP_ * D_))[j];
        }
    }
    __syncthreads();
    float gd = g[d], bd = bb[d];
    for (int c = 0; c < C_; ++c) {
        float h1 = slab[c][d];
        #pragma unroll
        for (int p = 0; p < TOPK_; ++p)
            h1 += wl[c][p] * slab[il[c][p]][d];
        float s = h1, s2 = h1 * h1;
        #pragma unroll
        for (int o = 16; o > 0; o >>= 1) {
            s  += __shfl_xor(s,  o, 32);
            s2 += __shfl_xor(s2, o, 32);
        }
        if (lane == 0) { red[0][wave] = s; red[1][wave] = s2; }
        __syncthreads();
        float ts = 0.f, ts2 = 0.f;
        #pragma unroll
        for (int w = 0; w < 8; ++w) { ts += red[0][w]; ts2 += red[1][w]; }
        float mu  = ts  * (1.0f / D_);
        float var = ts2 * (1.0f / D_) - mu * mu;
        float x = (h1 - mu) * rsqrtf(var + 1e-5f) * gd + bd;
        size_t ridx = ((size_t)(b * C_ + c) * NP_ + n) * D_ + d;
        H1out[ridx] = h1;
        X[ridx] = (__bf16)x;
        __syncthreads();
    }
}

// ---------------- K4: fused MLP (bf16 WMMA, fp32 accumulate), 64-row tiles, 16 waves ----------------
__global__ void __launch_bounds__(NT_)
k_mlp(const __bf16* __restrict__ X,
      const __bf16* __restrict__ W1T, const __bf16* __restrict__ W2T,
      const float* __restrict__ b1, const float* __restrict__ b2,
      float* __restrict__ out) {
    size_t row0 = (size_t)blockIdx.x * ROWS_;
    int t    = threadIdx.x;
    int lane = t & 31;
    int wave = t >> 5;             // 0..15
    int m    = lane & 15;
    int hi   = lane >> 4;          // 0/1

    __shared__ __align__(16) __bf16 xs[ROWS_][XPAD_];   // ~33 KB
    __shared__ __align__(16) __bf16 hs[ROWS_][HPAD_];   // ~65 KB (reused as fp32 out-stage)
    __shared__ __align__(16) __bf16 wp[D2_ * WPAD_];    // 40 KB transposed K-panel [n][k]

    // stage x tile: 64 rows x 256 bf16, coalesced
    {
        const uint4* src = (const uint4*)(X + row0 * D_);
        for (int idx = t; idx < ROWS_ * (D_ / 8); idx += NT_) {  // 2048 uint4
            int r = idx >> 5, j = idx & 31;
            *(uint4*)&xs[r][j * 8] = src[idx];
        }
    }
    __syncthreads();

    // ---- GEMM1: (64x256) @ (256x512), K-step 32 ----
    // 4 M-tiles x 32 N-tiles = 128 tiles / 16 waves = 8 tiles per wave (2M x 4N)
    int mtb1 = (wave >> 3) * 2;    // 0 or 2
    int ntb1 = (wave & 7) * 4;     // 0..28
    v8f acc[2][4] = {};
    for (int k0 = 0; k0 < D_; k0 += 32) {
        __syncthreads();
        { // stage transposed W1 panel: 512 n-rows x 32 k
            for (int idx = t; idx < D2_ * 4; idx += NT_) {       // 2048 uint4
                int n = idx >> 2, j = idx & 3;
                *(uint4*)&wp[n * WPAD_ + j * 8] =
                    *(const uint4*)(W1T + (size_t)n * D_ + k0 + j * 8);
            }
        }
        __syncthreads();
        Frag a[2];
        #pragma unroll
        for (int mt = 0; mt < 2; ++mt) {
            int row = (mtb1 + mt) * 16 + m;
            a[mt].q[0] = *(const uint4*)&xs[row][k0 + hi * 8];
            a[mt].q[1] = *(const uint4*)&xs[row][k0 + 16 + hi * 8];
        }
        #pragma unroll
        for (int j = 0; j < 4; ++j) {
            int col = (ntb1 + j) * 16 + m;
            Frag bfr;
            bfr.q[0] = *(const uint4*)&wp[col * WPAD_ + hi * 8];
            bfr.q[1] = *(const uint4*)&wp[col * WPAD_ + 16 + hi * 8];
            #pragma unroll
            for (int mt = 0; mt < 2; ++mt)
                acc[mt][j] = __builtin_amdgcn_wmma_f32_16x16x32_bf16(
                    false, a[mt].v, false, bfr.v, (short)0, acc[mt][j], false, false);
        }
    }
    // epilogue GEMM1: bias + exact gelu -> hs (bf16)
    __syncthreads();
    #pragma unroll
    for (int j = 0; j < 4; ++j) {
        int col = (ntb1 + j) * 16 + m;
        float bias = b1[col];
        #pragma unroll
        for (int mt = 0; mt < 2; ++mt) {
            #pragma unroll
            for (int r = 0; r < 8; ++r) {
                int mm = (mtb1 + mt) * 16 + r + hi * 8;
                float v = acc[mt][j][r] + bias;
                float ge = 0.5f * v * (1.0f + erff(v * 0.70710678118f));
                hs[mm][col] = (__bf16)ge;
            }
        }
    }
    __syncthreads();

    // ---- GEMM2: (64x512) @ (512x256), K-step 32 ----
    // 4 M-tiles x 16 N-tiles = 64 tiles / 16 waves = 4 tiles per wave (1M x 4N)
    int mt2  = wave >> 2;          // 0..3
    int ntb2 = (wave & 3) * 4;     // 0..12
    v8f acc2[4] = {};
    for (int k0 = 0; k0 < D2_; k0 += 32) {
        __syncthreads();
        { // stage transposed W2 panel: 256 n-rows x 32 k
            for (int idx = t; idx < D_ * 4; idx += NT_) {        // 1024 uint4
                int n = idx >> 2, j = idx & 3;
                *(uint4*)&wp[n * WPAD_ + j * 8] =
                    *(const uint4*)(W2T + (size_t)n * D2_ + k0 + j * 8);
            }
        }
        __syncthreads();
        Frag a;
        {
            int row = mt2 * 16 + m;
            a.q[0] = *(const uint4*)&hs[row][k0 + hi * 8];
            a.q[1] = *(const uint4*)&hs[row][k0 + 16 + hi * 8];
        }
        #pragma unroll
        for (int j = 0; j < 4; ++j) {
            int col = (ntb2 + j) * 16 + m;
            Frag bfr;
            bfr.q[0] = *(const uint4*)&wp[col * WPAD_ + hi * 8];
            bfr.q[1] = *(const uint4*)&wp[col * WPAD_ + 16 + hi * 8];
            acc2[j] = __builtin_amdgcn_wmma_f32_16x16x32_bf16(
                false, a.v, false, bfr.v, (short)0, acc2[j], false, false);
        }
    }
    // epilogue GEMM2: stage fp32 results in LDS (reuse hs), then coalesced RMW of out
    __syncthreads();               // hs (bf16) dead from here
    float* fo = (float*)&hs[0][0]; // 64 x 256 fp32 = 64 KB
    #pragma unroll
    for (int j = 0; j < 4; ++j) {
        int col = (ntb2 + j) * 16 + m;
        float bias = b2[col];
        #pragma unroll
        for (int r = 0; r < 8; ++r) {
            int mm = mt2 * 16 + r + hi * 8;
            fo[mm * D_ + col] = acc2[j][r] + bias;
        }
    }
    __syncthreads();
    { // out = H1 (already in out) + staged MLP result, coalesced float4
        float4* outv = (float4*)(out + row0 * D_);
        const float4* fov = (const float4*)fo;
        for (int i = t; i < ROWS_ * (D_ / 4); i += NT_) {  // 4096 float4
            float4 o = outv[i];
            float4 v = fov[i];
            o.x += v.x; o.y += v.y; o.z += v.z; o.w += v.w;
            outv[i] = o;
        }
    }
}

// ---------------- launch ----------------
extern "C" void kernel_launch(void* const* d_in, const int* in_sizes, int n_in,
                              void* d_out, int out_size, void* d_ws, size_t ws_size,
                              hipStream_t stream) {
    (void)in_sizes; (void)n_in; (void)out_size; (void)ws_size;
    const float* H    = (const float*)d_in[0];
    const float* Wq   = (const float*)d_in[1];
    const float* bq   = (const float*)d_in[2];
    const float* Wk   = (const float*)d_in[3];
    const float* bk   = (const float*)d_in[4];
    const float* ln_g = (const float*)d_in[5];
    const float* ln_b = (const float*)d_in[6];
    const float* W1   = (const float*)d_in[7];
    const float* b1   = (const float*)d_in[8];
    const float* W2   = (const float*)d_in[9];
    const float* b2   = (const float*)d_in[10];
    float* out = (float*)d_out;

    char* ws = (char*)d_ws;
    float*  Hc  = (float*)(ws + ((size_t)0    << 10));
    float*  Q   = (float*)(ws + ((size_t)512  << 10));
    float*  K   = (float*)(ws + ((size_t)1024 << 10));
    float*  w8  = (float*)(ws + ((size_t)1536 << 10));
    int*    i8  = (int*)  (ws + ((size_t)1552 << 10));
    __bf16* W1T = (__bf16*)(ws + ((size_t)1568 << 10));
    __bf16* W2T = (__bf16*)(ws + ((size_t)1824 << 10));
    __bf16* X   = (__bf16*)(ws + ((size_t)2080 << 10));

    k_cvt   <<<(D_ * D2_ + 255) / 256, 256, 0, stream>>>(W1, W2, W1T, W2T);
    k_mean  <<<B_ * C_,  D_,  0, stream>>>(H, Hc);
    k_qk    <<<B_ * C_,  D_,  0, stream>>>(Hc, Wq, bq, Wk, bk, Q, K);
    k_attn  <<<B_,       256, 0, stream>>>(Q, K, w8, i8);
    k_mix_ln<<<B_ * NP_, D_,  0, stream>>>(H, w8, i8, ln_g, ln_b, out, X);
    k_mlp   <<<(B_ * C_ * NP_) / ROWS_, NT_, 0, stream>>>(X, W1T, W2T, b1, b2, out);
}
